// CoRTLayer_20658792694403
// MI455X (gfx1250) — compile-verified, matrix operand order checked by reading, not run on hardware
//
#include <hip/hip_runtime.h>
#include <hip/hip_bf16.h>
#include <math.h>

#define DEV static __device__ __forceinline__

typedef __attribute__((ext_vector_type(16))) __bf16 v16bf;
typedef __attribute__((ext_vector_type(8)))  __bf16 v8bf;
typedef __attribute__((ext_vector_type(8)))  float  v8f;

constexpr int BB = 4, LL = 2048, DD = 768, HH = 12, FF = 3072;
constexpr int MM = BB * LL;           // 8192 tokens
constexpr int KTOP = 307;             // round(0.15 * 2048)

DEV unsigned short f2bf(float f) {
  unsigned u = __float_as_uint(f);
  u += 0x7FFFu + ((u >> 16) & 1u);    // round-to-nearest-even
  return (unsigned short)(u >> 16);
}

DEV v16bf frag16(const unsigned short* p0, const unsigned short* p1) {
  v8bf lo = *(const v8bf*)p0;
  v8bf hi = *(const v8bf*)p1;
  return __builtin_shufflevector(lo, hi, 0,1,2,3,4,5,6,7,8,9,10,11,12,13,14,15);
}

DEV v8f vzero8() {
  v8f z;
  #pragma unroll
  for (int i = 0; i < 8; i++) z[i] = 0.f;
  return z;
}

DEV float wsum32(float v) { for (int m = 16; m >= 1; m >>= 1) v += __shfl_xor(v, m, 32); return v; }
DEV float hsum16(float v) { for (int m = 8;  m >= 1; m >>= 1) v += __shfl_xor(v, m, 32); return v; }
DEV float hmax16(float v) { for (int m = 8;  m >= 1; m >>= 1) v = fmaxf(v, __shfl_xor(v, m, 32)); return v; }

// ---------------- LayerNorm1 + router score --------------------------------
__global__ __launch_bounds__(256) void k_ln_router(
    const float* __restrict__ x, const float* __restrict__ g, const float* __restrict__ bv,
    const float* __restrict__ Wr, const float* __restrict__ br,
    float* __restrict__ nf, unsigned short* __restrict__ nb, float* __restrict__ sc)
{
  int tok = blockIdx.x, tid = threadIdx.x;
  const float* xr = x + (size_t)tok * DD;
  float v0 = xr[tid], v1 = xr[tid + 256], v2 = xr[tid + 512];
  __shared__ float red[8];
  __shared__ float st[2];
  float s = wsum32(v0 + v1 + v2);
  if ((tid & 31) == 0) red[tid >> 5] = s;
  __syncthreads();
  if (tid == 0) { float t = 0; for (int i = 0; i < 8; i++) t += red[i]; st[0] = t * (1.0f / DD); }
  __syncthreads();
  float mean = st[0];
  float d0 = v0 - mean, d1 = v1 - mean, d2 = v2 - mean;
  float ss = wsum32(d0 * d0 + d1 * d1 + d2 * d2);
  __syncthreads();
  if ((tid & 31) == 0) red[tid >> 5] = ss;
  __syncthreads();
  if (tid == 0) { float t = 0; for (int i = 0; i < 8; i++) t += red[i]; st[1] = rsqrtf(t * (1.0f / DD) + 1e-5f); }
  __syncthreads();
  float rstd = st[1];
  float n0 = d0 * rstd * g[tid]       + bv[tid];
  float n1 = d1 * rstd * g[tid + 256] + bv[tid + 256];
  float n2 = d2 * rstd * g[tid + 512] + bv[tid + 512];
  size_t base = (size_t)tok * DD;
  nf[base + tid] = n0; nf[base + tid + 256] = n1; nf[base + tid + 512] = n2;
  nb[base + tid] = f2bf(n0); nb[base + tid + 256] = f2bf(n1); nb[base + tid + 512] = f2bf(n2);
  float dot = wsum32(n0 * Wr[tid] + n1 * Wr[tid + 256] + n2 * Wr[tid + 512]);
  __syncthreads();
  if ((tid & 31) == 0) red[tid >> 5] = dot;
  __syncthreads();
  if (tid == 0) {
    float t = 0; for (int i = 0; i < 8; i++) t += red[i];
    t += br[0];
    sc[tok] = 1.0f / (1.0f + __expf(-t));
  }
}

// ---------------- top-k mask via rank counting -----------------------------
__global__ __launch_bounds__(256) void k_mask(const float* __restrict__ sc, int* __restrict__ mask)
{
  int tok = blockIdx.x * 256 + threadIdx.x;
  int b = tok / LL, i = tok % LL;
  float si = sc[tok];
  const float* sb = sc + (size_t)b * LL;
  int cnt = 0;
  for (int j = 0; j < LL; j++) {
    float sj = sb[j];
    cnt += (sj < si) || ((sj == si) && (j < i));
  }
  mask[tok] = (cnt < KTOP) ? 1 : 0;
}

// ---------------- per-batch score sum + pooled -----------------------------
__global__ __launch_bounds__(256) void k_sums(const float* __restrict__ sc, float* __restrict__ sums)
{
  int b = blockIdx.x, tid = threadIdx.x;
  float s = 0;
  for (int l = tid; l < LL; l += 256) s += sc[(size_t)b * LL + l];
  s = wsum32(s);
  __shared__ float red[8];
  if ((tid & 31) == 0) red[tid >> 5] = s;
  __syncthreads();
  if (tid == 0) { float t = 0; for (int i = 0; i < 8; i++) t += red[i]; sums[b] = t; }
}

__global__ __launch_bounds__(256) void k_pooled(
    const float* __restrict__ sc, const float* __restrict__ nf,
    const float* __restrict__ sums, float* __restrict__ pooled)
{
  int b = blockIdx.y;
  int d = blockIdx.x * 256 + threadIdx.x;
  float acc = 0;
  for (int l = 0; l < LL; l++) acc += sc[(size_t)b * LL + l] * nf[((size_t)b * LL + l) * DD + d];
  pooled[(size_t)b * DD + d] = acc / (sums[b] + 1e-6f);
}

__global__ __launch_bounds__(256) void k_mixin(
    const float* __restrict__ nf, const float* __restrict__ sc,
    const float* __restrict__ pooled, unsigned short* __restrict__ y)
{
  size_t idx = (size_t)blockIdx.x * 256 + threadIdx.x;
  int tok = (int)(idx / DD), d = (int)(idx % DD);
  int b = tok / LL;
  y[idx] = f2bf(nf[idx] + sc[tok] * pooled[(size_t)b * DD + d]);
}

// ---------------- WMMA GEMM: C[M,N] = A_bf16[M,K] @ W_f32[K,N] + bias ------
// Double-buffered LDS staging: stage chunk kc+1 while WMMAs consume chunk kc.
DEV void gemm_stage(const unsigned short* __restrict__ A, const float* __restrict__ W,
                    unsigned short* sA, unsigned short* sW,
                    int tid, int m0, int n0, int k0, int Kdim, int N)
{
  { // A tile: 128x32 bf16, 16 elems/thread
    int row = tid >> 1, half = tid & 1;
    const uint4* src = (const uint4*)(A + (size_t)(m0 + row) * Kdim + k0 + half * 16);
    uint4* dst = (uint4*)(sA + row * 32 + half * 16);
    dst[0] = src[0]; dst[1] = src[1];
  }
  { // W tile transposed: 32x64 f32 -> bf16 [n][k], 8 elems/thread
    int k = tid >> 3, seg = tid & 7;
    const float4* src = (const float4*)(W + (size_t)(k0 + k) * N + n0 + seg * 8);
    float4 f0 = src[0], f1 = src[1];
    int nb = seg * 8;
    sW[(nb + 0) * 40 + k] = f2bf(f0.x); sW[(nb + 1) * 40 + k] = f2bf(f0.y);
    sW[(nb + 2) * 40 + k] = f2bf(f0.z); sW[(nb + 3) * 40 + k] = f2bf(f0.w);
    sW[(nb + 4) * 40 + k] = f2bf(f1.x); sW[(nb + 5) * 40 + k] = f2bf(f1.y);
    sW[(nb + 6) * 40 + k] = f2bf(f1.z); sW[(nb + 7) * 40 + k] = f2bf(f1.w);
  }
}

// MODE 0: f32 out | 1: bf16 out | 2: bf16 out w/ exact GELU | 3: f32 out + resid
template<int MODE>
__global__ __launch_bounds__(256) void k_gemm(
    const unsigned short* __restrict__ A, const float* __restrict__ W,
    const float* __restrict__ bias, const float* __restrict__ resid,
    void* __restrict__ OutV, int Kdim, int N)
{
  __shared__ unsigned short ldsA[2][128 * 32];   // [row][k], stride 32
  __shared__ unsigned short ldsW[2][64 * 40];    // transposed [n][k], stride 40
  int tid = threadIdx.x, lane = tid & 31, wid = tid >> 5;
  int wm = wid >> 1, wn = wid & 1;
  int m0 = blockIdx.x * 128, n0 = blockIdx.y * 64;
  v8f acc[2][2];
  #pragma unroll
  for (int i = 0; i < 2; i++) { acc[i][0] = vzero8(); acc[i][1] = vzero8(); }

  int nchunks = Kdim >> 5;
  int ml = lane & 15, hi = lane >> 4;

  gemm_stage(A, W, ldsA[0], ldsW[0], tid, m0, n0, 0, Kdim, N);
  __syncthreads();

  for (int kc = 0; kc < nchunks; kc++) {
    int cur = kc & 1;
    if (kc + 1 < nchunks)
      gemm_stage(A, W, ldsA[cur ^ 1], ldsW[cur ^ 1], tid, m0, n0, (kc + 1) << 5, Kdim, N);
    if (kc + 2 < nchunks) { // pull chunk kc+2 toward L2 (global_prefetch_b8)
      int k2 = (kc + 2) << 5;
      __builtin_prefetch(A + (size_t)(m0 + (tid >> 1)) * Kdim + k2, 0, 0);
      __builtin_prefetch(W + (size_t)(k2 + (tid >> 3)) * N + n0, 0, 0);
    }
    v16bf af[2], bfr[2];
    #pragma unroll
    for (int i = 0; i < 2; i++) {
      int r = wm * 32 + i * 16 + ml;
      af[i] = frag16(&ldsA[cur][r * 32 + hi * 8], &ldsA[cur][r * 32 + 16 + hi * 8]);
    }
    #pragma unroll
    for (int j = 0; j < 2; j++) {
      int c = wn * 32 + j * 16 + ml;
      bfr[j] = frag16(&ldsW[cur][c * 40 + hi * 8], &ldsW[cur][c * 40 + 16 + hi * 8]);
    }
    #pragma unroll
    for (int i = 0; i < 2; i++)
      #pragma unroll
      for (int j = 0; j < 2; j++)
        acc[i][j] = __builtin_amdgcn_wmma_f32_16x16x32_bf16(
            false, af[i], false, bfr[j], (short)0, acc[i][j], false, false);
    __syncthreads();
  }
  #pragma unroll
  for (int i = 0; i < 2; i++)
    #pragma unroll
    for (int j = 0; j < 2; j++) {
      int col = n0 + wn * 32 + j * 16 + ml;
      float bcol = bias[col];
      #pragma unroll
      for (int r = 0; r < 8; r++) {
        int row = m0 + wm * 32 + i * 16 + 8 * hi + r;
        float vsum = acc[i][j][r] + bcol;
        size_t oi = (size_t)row * N + col;
        if (MODE == 0) {
          ((float*)OutV)[oi] = vsum;
        } else if (MODE == 1) {
          ((unsigned short*)OutV)[oi] = f2bf(vsum);
        } else if (MODE == 2) {
          float gel = 0.5f * vsum * (1.0f + erff(vsum * 0.70710678118654752f));
          ((unsigned short*)OutV)[oi] = f2bf(gel);
        } else {
          ((float*)OutV)[oi] = resid[oi] + vsum;
        }
      }
    }
}

// ---------------- Flash attention (WMMA, streaming softmax) ----------------
DEV void attn_stage_kv(const unsigned short* __restrict__ Kb,
                       const unsigned short* __restrict__ Vb,
                       unsigned short* sK, unsigned short* sV,
                       int tid, size_t basekv /* (b*LL+k0)*DD + h*64 */)
{
  int key = tid >> 2, seg = tid & 3;
  size_t go = basekv + (size_t)key * DD + seg * 16;
  const uint4* ks = (const uint4*)(Kb + go);
  uint4* kd = (uint4*)(sK + key * 72 + seg * 16);
  kd[0] = ks[0]; kd[1] = ks[1];
  const uint4* vs = (const uint4*)(Vb + go);
  uint4 a0 = vs[0], a1 = vs[1];
  unsigned wv[8] = {a0.x, a0.y, a0.z, a0.w, a1.x, a1.y, a1.z, a1.w};
  #pragma unroll
  for (int i = 0; i < 8; i++) {
    int d = seg * 16 + 2 * i;
    sV[d * 40 + key]       = (unsigned short)(wv[i] & 0xffffu);
    sV[(d + 1) * 40 + key] = (unsigned short)(wv[i] >> 16);
  }
}

__global__ __launch_bounds__(128) void k_attn(
    const unsigned short* __restrict__ Qb, const unsigned short* __restrict__ Kb,
    const unsigned short* __restrict__ Vb, unsigned short* __restrict__ Ctx)
{
  __shared__ unsigned short ldsQ[64 * 72];        // [qrow][d], stride 72
  __shared__ unsigned short ldsK[2][32 * 72];     // [key][d],  stride 72 (double-buffered)
  __shared__ unsigned short ldsV[2][64 * 40];     // transposed [d][key], stride 40
  __shared__ unsigned short ldsP[4][16 * 40];     // per-wave P tile [m][key]
  int tid = threadIdx.x, lane = tid & 31, wid = tid >> 5;
  int b = blockIdx.z, h = blockIdx.y, qt = blockIdx.x;
  size_t qrow0 = (size_t)b * LL + qt * 64;
  size_t bh = (size_t)b * LL * DD + (size_t)h * 64;

  if (tid < 64) { // stage all 64 Q rows of this tile group
    const uint4* src = (const uint4*)(Qb + (qrow0 + tid) * DD + h * 64);
    uint4* dst = (uint4*)(&ldsQ[tid * 72]);
    #pragma unroll
    for (int i = 0; i < 8; i++) dst[i] = src[i];
  }
  attn_stage_kv(Kb, Vb, ldsK[0], ldsV[0], tid, bh);
  __syncthreads();

  int nl = lane & 15, hi = lane >> 4;
  v16bf qf[2];
  #pragma unroll
  for (int c = 0; c < 2; c++) {
    int r = wid * 16 + nl;
    qf[c] = frag16(&ldsQ[r * 72 + c * 32 + hi * 8], &ldsQ[r * 72 + c * 32 + 16 + hi * 8]);
  }
  float mrow[8], lrow[8];
  v8f acc[4];
  #pragma unroll
  for (int r = 0; r < 8; r++) { mrow[r] = -1e30f; lrow[r] = 0.f; }
  #pragma unroll
  for (int jn = 0; jn < 4; jn++) acc[jn] = vzero8();

  constexpr int NCH = LL / 32;
  for (int kc = 0; kc < NCH; kc++) {
    int cur = kc & 1;
    if (kc + 1 < NCH)
      attn_stage_kv(Kb, Vb, ldsK[cur ^ 1], ldsV[cur ^ 1], tid, bh + (size_t)(kc + 1) * 32 * DD);
    if (kc + 2 < NCH) {
      size_t p = bh + (size_t)(kc + 2) * 32 * DD + (size_t)(tid >> 2) * DD;
      __builtin_prefetch(Kb + p, 0, 0);
      __builtin_prefetch(Vb + p, 0, 0);
    }
    // S = Q (16x64) @ K^T (64x32) -> two 16x16 tiles
    v8f sacc[2];
    #pragma unroll
    for (int j = 0; j < 2; j++) {
      sacc[j] = vzero8();
      #pragma unroll
      for (int c = 0; c < 2; c++) {
        int kk = j * 16 + nl;
        v16bf kf = frag16(&ldsK[cur][kk * 72 + c * 32 + hi * 8],
                          &ldsK[cur][kk * 72 + c * 32 + 16 + hi * 8]);
        sacc[j] = __builtin_amdgcn_wmma_f32_16x16x32_bf16(
            false, qf[c], false, kf, (short)0, sacc[j], false, false);
      }
    }
    // online softmax row update
    float p0[8], p1[8];
    #pragma unroll
    for (int r = 0; r < 8; r++) {
      float s0 = sacc[0][r] * 0.125f;   // 1/sqrt(64)
      float s1 = sacc[1][r] * 0.125f;
      float mx = hmax16(fmaxf(s0, s1));
      float mnew = fmaxf(mrow[r], mx);
      float alpha = __expf(mrow[r] - mnew);
      p0[r] = __expf(s0 - mnew);
      p1[r] = __expf(s1 - mnew);
      float rs = hsum16(p0[r] + p1[r]);
      lrow[r] = lrow[r] * alpha + rs;
      mrow[r] = mnew;
      #pragma unroll
      for (int jn = 0; jn < 4; jn++) acc[jn][r] *= alpha;
    }
    // stage P (C-layout -> LDS) then reload as A-fragment
    #pragma unroll
    for (int r = 0; r < 8; r++) {
      int rowp = r + 8 * hi;
      ldsP[wid][rowp * 40 + nl]      = f2bf(p0[r]);
      ldsP[wid][rowp * 40 + 16 + nl] = f2bf(p1[r]);
    }
    __syncthreads();
    v16bf pf = frag16(&ldsP[wid][nl * 40 + hi * 8], &ldsP[wid][nl * 40 + 16 + hi * 8]);
    #pragma unroll
    for (int jn = 0; jn < 4; jn++) {
      int d = jn * 16 + nl;
      v16bf vf = frag16(&ldsV[cur][d * 40 + hi * 8], &ldsV[cur][d * 40 + 16 + hi * 8]);
      acc[jn] = __builtin_amdgcn_wmma_f32_16x16x32_bf16(
          false, pf, false, vf, (short)0, acc[jn], false, false);
    }
    __syncthreads();   // protect ldsV[cur]/ldsK[cur] before re-staging next iter
  }
  // epilogue: ctx = acc / l
  #pragma unroll
  for (int jn = 0; jn < 4; jn++)
    #pragma unroll
    for (int r = 0; r < 8; r++) {
      float o = acc[jn][r] / lrow[r];
      size_t row = qrow0 + wid * 16 + 8 * hi + r;
      Ctx[row * DD + h * 64 + jn * 16 + nl] = f2bf(o);
    }
}

// ---------------- merge (route) + residual + LayerNorm2 --------------------
__global__ __launch_bounds__(256) void k_merge_ln2(
    const float* __restrict__ x, const int* __restrict__ mask,
    const float* __restrict__ attn, const float* __restrict__ mixo,
    const float* __restrict__ g, const float* __restrict__ bv,
    float* __restrict__ hout, unsigned short* __restrict__ n2b)
{
  int tok = blockIdx.x, tid = threadIdx.x;
  size_t base = (size_t)tok * DD;
  int m = mask[tok];
  float v[3];
  #pragma unroll
  for (int i = 0; i < 3; i++) {
    size_t idx = base + tid + i * 256;
    float mrg = m ? attn[idx] : mixo[idx];
    v[i] = x[idx] + mrg;
    hout[idx] = v[i];
  }
  __shared__ float red[8];
  __shared__ float st[2];
  float s = wsum32(v[0] + v[1] + v[2]);
  if ((tid & 31) == 0) red[tid >> 5] = s;
  __syncthreads();
  if (tid == 0) { float t = 0; for (int i = 0; i < 8; i++) t += red[i]; st[0] = t * (1.0f / DD); }
  __syncthreads();
  float mean = st[0];
  float d0 = v[0] - mean, d1 = v[1] - mean, d2 = v[2] - mean;
  float ss = wsum32(d0 * d0 + d1 * d1 + d2 * d2);
  __syncthreads();
  if ((tid & 31) == 0) red[tid >> 5] = ss;
  __syncthreads();
  if (tid == 0) { float t = 0; for (int i = 0; i < 8; i++) t += red[i]; st[1] = rsqrtf(t * (1.0f / DD) + 1e-5f); }
  __syncthreads();
  float rstd = st[1];
  n2b[base + tid]       = f2bf(d0 * rstd * g[tid]       + bv[tid]);
  n2b[base + tid + 256] = f2bf(d1 * rstd * g[tid + 256] + bv[tid + 256]);
  n2b[base + tid + 512] = f2bf(d2 * rstd * g[tid + 512] + bv[tid + 512]);
}

// ---------------- driver ---------------------------------------------------
extern "C" void kernel_launch(void* const* d_in, const int* in_sizes, int n_in,
                              void* d_out, int out_size, void* d_ws, size_t ws_size,
                              hipStream_t stream)
{
  (void)in_sizes; (void)n_in; (void)out_size; (void)ws_size;
  const float* x  = (const float*)d_in[0];
  const float* g1 = (const float*)d_in[1];  const float* b1 = (const float*)d_in[2];
  const float* g2 = (const float*)d_in[3];  const float* b2 = (const float*)d_in[4];
  const float* Wr = (const float*)d_in[5];  const float* br = (const float*)d_in[6];
  const float* Wq = (const float*)d_in[7];  const float* bq = (const float*)d_in[8];
  const float* Wk = (const float*)d_in[9];  const float* bk = (const float*)d_in[10];
  const float* Wv = (const float*)d_in[11]; const float* bv = (const float*)d_in[12];
  const float* Wo = (const float*)d_in[13]; const float* bo = (const float*)d_in[14];
  const float* Wm = (const float*)d_in[15]; const float* bm = (const float*)d_in[16];
  const float* W1 = (const float*)d_in[17]; const float* bf1 = (const float*)d_in[18];
  const float* W2 = (const float*)d_in[19]; const float* bf2 = (const float*)d_in[20];

  char* ws = (char*)d_ws;
  size_t off = 0;
  auto alloc = [&](size_t bytes) -> char* {
    char* p = ws + off;
    off = (off + bytes + 255) & ~(size_t)255;
    return p;
  };
  float*          nf     = (float*)alloc((size_t)MM * DD * 4);
  unsigned short* nb     = (unsigned short*)alloc((size_t)MM * DD * 2);
  float*          sc     = (float*)alloc((size_t)MM * 4);
  int*            maskp  = (int*)alloc((size_t)MM * 4);
  float*          sums   = (float*)alloc(64);
  float*          pooled = (float*)alloc((size_t)BB * DD * 4);
  unsigned short* qb     = (unsigned short*)alloc((size_t)MM * DD * 2);
  unsigned short* kb     = (unsigned short*)alloc((size_t)MM * DD * 2);
  unsigned short* vb     = (unsigned short*)alloc((size_t)MM * DD * 2);
  unsigned short* ctx    = (unsigned short*)alloc((size_t)MM * DD * 2);
  float*          attn   = (float*)alloc((size_t)MM * DD * 4);
  unsigned short* mixin  = (unsigned short*)alloc((size_t)MM * DD * 2);
  float*          mixo   = (float*)alloc((size_t)MM * DD * 4);
  float*          hbuf   = (float*)alloc((size_t)MM * DD * 4);
  unsigned short* n2b    = (unsigned short*)alloc((size_t)MM * DD * 2);
  unsigned short* f1b    = (unsigned short*)alloc((size_t)MM * FF * 2);

  k_ln_router<<<MM, 256, 0, stream>>>(x, g1, b1, Wr, br, nf, nb, sc);
  k_mask<<<MM / 256, 256, 0, stream>>>(sc, maskp);
  k_sums<<<BB, 256, 0, stream>>>(sc, sums);
  k_pooled<<<dim3(DD / 256, BB), 256, 0, stream>>>(sc, nf, sums, pooled);

  dim3 gqkv(MM / 128, DD / 64);
  k_gemm<1><<<gqkv, 256, 0, stream>>>(nb, Wq, bq, nullptr, qb, DD, DD);
  k_gemm<1><<<gqkv, 256, 0, stream>>>(nb, Wk, bk, nullptr, kb, DD, DD);
  k_gemm<1><<<gqkv, 256, 0, stream>>>(nb, Wv, bv, nullptr, vb, DD, DD);

  k_attn<<<dim3(LL / 64, HH, BB), 128, 0, stream>>>(qb, kb, vb, ctx);

  k_gemm<0><<<gqkv, 256, 0, stream>>>(ctx, Wo, bo, nullptr, attn, DD, DD);
  k_mixin<<<(MM * DD) / 256, 256, 0, stream>>>(nf, sc, pooled, mixin);
  k_gemm<0><<<gqkv, 256, 0, stream>>>(mixin, Wm, bm, nullptr, mixo, DD, DD);

  k_merge_ln2<<<MM, 256, 0, stream>>>(x, maskp, attn, mixo, g2, b2, hbuf, n2b);

  k_gemm<2><<<dim3(MM / 128, FF / 64), 256, 0, stream>>>(n2b, W1, bf1, nullptr, f1b, DD, FF);
  k_gemm<3><<<dim3(MM / 128, DD / 64), 256, 0, stream>>>(f1b, W2, bf2, hbuf, d_out, FF, DD);
}
